// MultiheadSelfAttention_50105088475585
// MI455X (gfx1250) — compile-verified, matrix operand order checked by reading, not run on hardware
//
#include <hip/hip_runtime.h>
#include <hip/hip_bf16.h>

// MI455X (gfx1250) multi-head causal self-attention, bf16 WMMA path.
// GEMMs: TDM (tensor_load_to_lds, TENSORcnt) double-buffered tile pipeline.
// Attention: flash-style, async-to-LDS V staging (ASYNCcnt) + prefetch.
// B=2, S=4096, D=1024, H=16, hd=64.

typedef __attribute__((ext_vector_type(16))) __bf16  v16bf;
typedef __attribute__((ext_vector_type(8)))  __bf16  v8bf;
typedef __attribute__((ext_vector_type(8)))  float   v8f;
typedef __attribute__((ext_vector_type(4)))  uint32_t v4u;
typedef __attribute__((ext_vector_type(8)))  uint32_t v8u;

#define Bdim 2
#define Sdim 4096
#define Ddim 1024
#define Hdim 16
#define HD   64
#define Mtot (Bdim * Sdim)   // 8192 token rows

static __device__ __forceinline__ v8f vzero8() {
  v8f z = {0.f, 0.f, 0.f, 0.f, 0.f, 0.f, 0.f, 0.f};
  return z;
}

// D = A(16x32 bf16) * B(32x16 bf16) + C, f32 accumulate.
static __device__ __forceinline__ v8f wmma_bf16(v16bf a, v16bf b, v8f c) {
  return __builtin_amdgcn_wmma_f32_16x16x32_bf16(
      /*neg_a=*/false, a, /*neg_b=*/false, b,
      /*c_mod=*/(short)0, c, /*reuse_a=*/false, /*reuse_b=*/false);
}

// ---- ASYNCcnt path: 16B global -> LDS copy, no VGPR data movement. --------
static __device__ __forceinline__ void async_cp16(void* ldsdst,
                                                  const void* gsrc) {
  const uint32_t ldso = (uint32_t)(uintptr_t)ldsdst;  // LDS byte offset
  asm volatile("global_load_async_to_lds_b128 %0, %1, off"
               :: "v"(ldso), "v"(gsrc)
               : "memory");
}
static __device__ __forceinline__ void wait_async0() {
  asm volatile("s_wait_asynccnt 0x0" ::: "memory");
}

// ---- TENSORcnt path: TDM load of a 128x32 bf16 tile into padded LDS. ------
// D# per ISA cdna5 ch.8: group0 {count=1, lds_addr, global_addr, type=2},
// group1 {data_size=2B, pad_enable, pad_interval=16DW (one 64B row),
// pad_amount=8DW (32B) -> LDS row stride 48 elems, dims/strides}, groups
// 2/3 zero (2-D tile). Descriptors are wave-uniform -> "s" SGPR groups.
static __device__ __forceinline__ void tdm_load_tile_128x32(
    void* ldsdst, const __bf16* gsrc) {
  const uint64_t ga = (uint64_t)(uintptr_t)gsrc;
  v4u g0;
  g0[0] = 1u;                                   // count=1 (valid user D#)
  g0[1] = (uint32_t)(uintptr_t)ldsdst;          // lds_addr (bytes)
  g0[2] = (uint32_t)ga;                         // global_addr[31:0]
  g0[3] = (uint32_t)((ga >> 32) & 0x1FFFFFFu)   // global_addr[56:32]
        | (2u << 30);                           // type = 2 ("image")
  v8u g1;
  g1[0] = (1u << 16)                            // data_size: 2 bytes
        | (1u << 20)                            // pad_enable
        | (3u << 22)                            // pad_interval: 16 DWORDs
        | (7u << 25);                           // pad_amount: 8 DWORDs
  g1[1] = (32u & 0xFFFFu) << 16;                // tensor_dim0 = 32 (lo16)
  g1[2] = 0u | ((128u & 0xFFFFu) << 16);        // dim0 hi | tensor_dim1 = 128
  g1[3] = 0u | (32u << 16);                     // dim1 hi | tile_dim0 = 32
  g1[4] = 128u;                                 // tile_dim1 = 128, tile_dim2=0
  g1[5] = (uint32_t)Ddim;                       // tensor_dim0_stride = 1024
  g1[6] = 0u;
  g1[7] = 0u;
  const v4u gz = {0u, 0u, 0u, 0u};
  asm volatile("tensor_load_to_lds %0, %1, %2, %3"
               :: "s"(g0), "s"(g1), "s"(gz), "s"(gz)
               : "memory");
}

// Fragment loader: per lane, 16 contiguous bf16 from row (lane%16),
// K-offset (lane/16)*16. 32B aligned by construction everywhere it is used.
static __device__ __forceinline__ v16bf ld_frag(const __bf16* rowbase,
                                                int stride, int lane) {
  const __bf16* p = rowbase + (size_t)(lane & 15) * stride + ((lane >> 4) << 4);
  return *(const v16bf*)p;
}

// Reductions across the 16 lanes of one wave half (wave32; masks < 16 stay
// inside a half). Each half holds one set of 16 columns of the C fragment.
static __device__ __forceinline__ float halfmax(float x) {
  x = fmaxf(x, __shfl_xor(x, 1, 32));
  x = fmaxf(x, __shfl_xor(x, 2, 32));
  x = fmaxf(x, __shfl_xor(x, 4, 32));
  x = fmaxf(x, __shfl_xor(x, 8, 32));
  return x;
}
static __device__ __forceinline__ float halfsum(float x) {
  x += __shfl_xor(x, 1, 32);
  x += __shfl_xor(x, 2, 32);
  x += __shfl_xor(x, 4, 32);
  x += __shfl_xor(x, 8, 32);
  return x;
}

// ---------------------------------------------------------------------------
// fp32 -> bf16 conversion (X and the four weight matrices), 8 elems/thread.
// ---------------------------------------------------------------------------
__global__ __launch_bounds__(256) void cvt_bf16_kernel(
    const float* __restrict__ in, __bf16* __restrict__ out, int n) {
  const int i = (blockIdx.x * 256 + threadIdx.x) * 8;
  if (i >= n) return;
  const float4 a = *(const float4*)(in + i);
  const float4 b = *(const float4*)(in + i + 4);
  v8bf o;
  o[0] = (__bf16)a.x; o[1] = (__bf16)a.y; o[2] = (__bf16)a.z; o[3] = (__bf16)a.w;
  o[4] = (__bf16)b.x; o[5] = (__bf16)b.y; o[6] = (__bf16)b.z; o[7] = (__bf16)b.w;
  *(v8bf*)(out + i) = o;
}

// ---------------------------------------------------------------------------
// GEMM: out[M x 1024] = A[M x 1024] * W^T, A and W bf16 row-major.
// 128x128 block tile, 256 threads = 8 waves (4x2 wave grid, 32x64 per wave,
// 2x4 WMMA tiles per 32-deep k-step). Double-buffered LDS fed by the Tensor
// Data Mover: wave 0 issues tensor_load_to_lds for tile k+1 while everyone
// computes tile k; one s_wait_tensorcnt + barrier per iteration.
// ---------------------------------------------------------------------------
template <typename TOUT>
__global__ __launch_bounds__(256) void gemm_wT_kernel(
    const __bf16* __restrict__ A, const __bf16* __restrict__ W,
    TOUT* __restrict__ out) {
  __shared__ __attribute__((aligned(32))) __bf16 ldsA[2][128 * 48];
  __shared__ __attribute__((aligned(32))) __bf16 ldsB[2][128 * 48];

  const int t = threadIdx.x;
  const int lane = t & 31;
  const int w = t >> 5;
  const int wm = w >> 1;          // 0..3
  const int wn = w & 1;           // 0..1
  const int m0 = blockIdx.x * 128;
  const int n0 = blockIdx.y * 128;
  constexpr int NT = Ddim / 32;   // 32 k-steps

  const __bf16* Abase = A + (size_t)m0 * Ddim;
  const __bf16* Wbase = W + (size_t)n0 * Ddim;

  v8f acc[2][4];
#pragma unroll
  for (int i = 0; i < 2; ++i)
#pragma unroll
    for (int j = 0; j < 4; ++j) acc[i][j] = vzero8();

  if (w == 0) {  // prologue: TDM streams tile 0
    tdm_load_tile_128x32(ldsA[0], Abase);
    tdm_load_tile_128x32(ldsB[0], Wbase);
  }

  for (int kt = 0; kt < NT; ++kt) {
    const int cur = kt & 1;
    __builtin_amdgcn_s_wait_tensorcnt(0);  // tile kt landed (wave0's tokens)
    __syncthreads();                       // visible to all; readers of the
                                           // other buffer are also done
    if (w == 0 && kt + 1 < NT) {           // TDM streams tile kt+1
      tdm_load_tile_128x32(ldsA[cur ^ 1], Abase + (kt + 1) * 32);
      tdm_load_tile_128x32(ldsB[cur ^ 1], Wbase + (kt + 1) * 32);
    }

    v16bf af[2], bfr[4];
#pragma unroll
    for (int i = 0; i < 2; ++i)
      af[i] = ld_frag(&ldsA[cur][(wm * 32 + i * 16) * 48], 48, lane);
#pragma unroll
    for (int j = 0; j < 4; ++j)
      bfr[j] = ld_frag(&ldsB[cur][(wn * 64 + j * 16) * 48], 48, lane);
#pragma unroll
    for (int i = 0; i < 2; ++i)
#pragma unroll
      for (int j = 0; j < 4; ++j)
        acc[i][j] = wmma_bf16(af[i], bfr[j], acc[i][j]);
  }

  // C/D fragment layout (ISA 7.12.2): n = lane%16, m = r + 8*(lane/16).
#pragma unroll
  for (int i = 0; i < 2; ++i)
#pragma unroll
    for (int j = 0; j < 4; ++j)
#pragma unroll
      for (int r = 0; r < 8; ++r) {
        const int row = m0 + wm * 32 + i * 16 + r + ((lane >> 4) << 3);
        const int col = n0 + wn * 64 + j * 16 + (lane & 15);
        out[(size_t)row * Ddim + col] = (TOUT)acc[i][j][r];
      }
}

// ---------------------------------------------------------------------------
// Flash attention, causal. Grid: (S/128, H, B), 256 threads = 8 waves.
// Each wave owns 16 query rows; loops over 32-key blocks up to the causal
// frontier. Scores: 2 k-steps (hd=64) x 2 key sub-tiles = 4 WMMAs.
// PV: P (16x32 bf16, re-laid-out via per-wave LDS) x V block = 4 WMMAs.
// V block staged with async-to-LDS; next K block prefetched.
// ---------------------------------------------------------------------------
__global__ __launch_bounds__(256) void flash_attn_kernel(
    const __bf16* __restrict__ Q, const __bf16* __restrict__ K,
    const __bf16* __restrict__ V, __bf16* __restrict__ Ab) {
  __shared__ __attribute__((aligned(32))) __bf16 ldsP[8 * 16 * 48];  // per-wave
  __shared__ __attribute__((aligned(32))) __bf16 ldsV[8 * 32 * 72];  // per-wave

  const int t = threadIdx.x;
  const int lane = t & 31;
  const int w = t >> 5;
  const int h = blockIdx.y;
  const int b = blockIdx.z;
  const int q0 = blockIdx.x * 128 + w * 16;  // first query row of this wave
  const int hh = lane >> 4;                  // fragment half
  const int ln = lane & 15;                  // fragment column

  const size_t baseTok = (size_t)b * Sdim;

  // Q fragments for this wave's 16 rows: hd=64 -> two 16x32 A fragments.
  const __bf16* qbase = Q + (baseTok + q0) * Ddim + h * HD;
  const v16bf qa0 = ld_frag(qbase, Ddim, lane);
  const v16bf qa1 = ld_frag(qbase + 32, Ddim, lane);

  float mrow[8], lrow[8];
  v8f o[4];
#pragma unroll
  for (int r = 0; r < 8; ++r) { mrow[r] = -1e30f; lrow[r] = 0.f; }
#pragma unroll
  for (int f = 0; f < 4; ++f) o[f] = vzero8();

  __bf16* pbuf = &ldsP[w * 16 * 48];
  __bf16* vbuf = &ldsV[w * 32 * 72];

  const int nblk = ((q0 + 15) >> 5) + 1;  // causal: keys <= q0+15
  for (int kb = 0; kb < nblk; ++kb) {
    const int key0 = kb * 32;

    // Stage V block (32 keys x 64) into per-wave LDS asynchronously:
    // one key row (128B = 8x16B) per lane.
    {
      const __bf16* vsrc = V + (baseTok + key0 + lane) * Ddim + h * HD;
      __bf16* vd = &vbuf[lane * 72];
#pragma unroll
      for (int sgm = 0; sgm < 8; ++sgm)
        async_cp16(&vd[sgm * 8], vsrc + sgm * 8);
    }

    // Scores = Q * K^T (K rows are K-contiguous -> B fragments load direct).
    const __bf16* k0p = K + (baseTok + key0) * Ddim + h * HD;
    const __bf16* k1p = k0p + (size_t)16 * Ddim;
    if (kb + 1 < nblk)  // prefetch next causal K block -> global_prefetch_b8
      __builtin_prefetch(k0p + (size_t)32 * Ddim, 0, 1);

    v8f s0 = vzero8(), s1 = vzero8();
    s0 = wmma_bf16(qa0, ld_frag(k0p, Ddim, lane), s0);
    s0 = wmma_bf16(qa1, ld_frag(k0p + 32, Ddim, lane), s0);
    s1 = wmma_bf16(qa0, ld_frag(k1p, Ddim, lane), s1);
    s1 = wmma_bf16(qa1, ld_frag(k1p + 32, Ddim, lane), s1);

    // Scale + causal mask (element (m,n): m = r+8*hh, n = ln).
    const float scale = 0.125f;  // 1/sqrt(64)
#pragma unroll
    for (int r = 0; r < 8; ++r) {
      const int qg = q0 + r + hh * 8;
      const int kg = key0 + ln;
      s0[r] = (kg <= qg) ? s0[r] * scale : -1e30f;
      s1[r] = (kg + 16 <= qg) ? s1[r] * scale : -1e30f;
    }

    // Online softmax (per row: 16 columns live across one wave half).
#pragma unroll
    for (int r = 0; r < 8; ++r) {
      const float rm = halfmax(fmaxf(s0[r], s1[r]));
      const float nm = fmaxf(mrow[r], rm);
      const float al = __expf(mrow[r] - nm);
      s0[r] = __expf(s0[r] - nm);
      s1[r] = __expf(s1[r] - nm);
      lrow[r] = lrow[r] * al + halfsum(s0[r] + s1[r]);
      mrow[r] = nm;
#pragma unroll
      for (int f = 0; f < 4; ++f) o[f][r] *= al;
    }

    // Re-layout P (C layout -> A fragment) through per-wave LDS as bf16.
#pragma unroll
    for (int r = 0; r < 8; ++r) {
      const int m = r + hh * 8;
      pbuf[m * 48 + ln] = (__bf16)s0[r];
      pbuf[m * 48 + 16 + ln] = (__bf16)s1[r];
    }
    // Per-wave ordering: DS stores -> DS loads (pbuf); async copies -> reads
    // (vbuf). Both counters are per-wave; regions are wave-private.
    asm volatile("s_wait_dscnt 0x0" ::: "memory");
    wait_async0();

    const v16bf pf = ld_frag(pbuf, 48, lane);

    // O += P * V ; V B-fragment is K(=key)-strided in LDS.
#pragma unroll
    for (int f = 0; f < 4; ++f) {
      v16bf vf;
      const __bf16* col = &vbuf[f * 16 + ln];
      const int kbase = hh * 16;
#pragma unroll
      for (int j = 0; j < 16; ++j) vf[j] = col[(kbase + j) * 72];
      o[f] = wmma_bf16(pf, vf, o[f]);
    }
  }

  // Normalize and store (bf16) for the output projection.
#pragma unroll
  for (int r = 0; r < 8; ++r) {
    const float rl = 1.0f / lrow[r];
    const int row = q0 + r + hh * 8;
#pragma unroll
    for (int f = 0; f < 4; ++f)
      Ab[(baseTok + row) * Ddim + h * HD + f * 16 + ln] =
          (__bf16)(o[f][r] * rl);
  }
}

// ---------------------------------------------------------------------------
extern "C" void kernel_launch(void* const* d_in, const int* in_sizes, int n_in,
                              void* d_out, int out_size, void* d_ws,
                              size_t ws_size, hipStream_t stream) {
  (void)in_sizes; (void)n_in; (void)out_size; (void)ws_size;
  const float* X  = (const float*)d_in[0];
  const float* Wq = (const float*)d_in[1];
  const float* Wk = (const float*)d_in[2];
  const float* Wv = (const float*)d_in[3];
  const float* Wo = (const float*)d_in[4];
  float* out = (float*)d_out;

  const size_t NE = (size_t)Mtot * Ddim;   // 8.39M elems (token tensors)
  const size_t WE = (size_t)Ddim * Ddim;   // 1.05M elems (weights)
  __bf16* Xb  = (__bf16*)d_ws;
  __bf16* Wqb = Xb + NE;
  __bf16* Wkb = Wqb + WE;
  __bf16* Wvb = Wkb + WE;
  __bf16* Wob = Wvb + WE;
  __bf16* Qb  = Wob + WE;
  __bf16* Kb  = Qb + NE;
  __bf16* Vb  = Kb + NE;
  __bf16* Ab  = Vb + NE;  // total ~92 MB of workspace

  // fp32 -> bf16 staging passes (traffic-trivial vs. ~138 GFLOP of WMMA).
  cvt_bf16_kernel<<<(int)(NE / 8 / 256), 256, 0, stream>>>(X, Xb, (int)NE);
  cvt_bf16_kernel<<<(int)(WE / 8 / 256), 256, 0, stream>>>(Wq, Wqb, (int)WE);
  cvt_bf16_kernel<<<(int)(WE / 8 / 256), 256, 0, stream>>>(Wk, Wkb, (int)WE);
  cvt_bf16_kernel<<<(int)(WE / 8 / 256), 256, 0, stream>>>(Wv, Wvb, (int)WE);
  cvt_bf16_kernel<<<(int)(WE / 8 / 256), 256, 0, stream>>>(Wo, Wob, (int)WE);

  const dim3 ggrid(Mtot / 128, Ddim / 128);  // 64 x 8
  gemm_wT_kernel<__bf16><<<ggrid, 256, 0, stream>>>(Xb, Wqb, Qb);
  gemm_wT_kernel<__bf16><<<ggrid, 256, 0, stream>>>(Xb, Wkb, Kb);
  gemm_wT_kernel<__bf16><<<ggrid, 256, 0, stream>>>(Xb, Wvb, Vb);

  const dim3 agrid(Sdim / 128, Hdim, Bdim);  // 32 x 16 x 2
  flash_attn_kernel<<<agrid, 256, 0, stream>>>(Qb, Kb, Vb, Ab);

  gemm_wT_kernel<float><<<ggrid, 256, 0, stream>>>(Ab, Wob, out);
}
